// TPTPEntailmentModel_70600672412126
// MI455X (gfx1250) — matrix operand order repaired and checked by reading, not compiled, not run on hardware
//
#include <hip/hip_runtime.h>
#include <hip/hip_bf16.h>
#include <stdint.h>

// ---------------------------------------------------------------------------
// GCN forward for MI455X (gfx1250, wave32, WMMA).
// Pipeline: deg -> bf16 convert/swizzle -> WMMA GEMM -> edge scatter (f32
// atomics, L2-resident hw) -> combine+ReLU -> repeat -> pool -> tiny MLP.
// ---------------------------------------------------------------------------

#define HID   256
#define GNUM  256

typedef __attribute__((ext_vector_type(16))) __bf16       v16bf;
typedef __attribute__((ext_vector_type(8)))  float        v8f;
typedef __attribute__((ext_vector_type(4)))  unsigned int uvec4;
typedef __attribute__((ext_vector_type(4)))  float        fvec4;

union ABFrag { uvec4 u[2]; v16bf v; };

__device__ __forceinline__ unsigned short f32_to_bf16(float f) {
  unsigned int u = __float_as_uint(f);
  unsigned int r = u + 0x7fffu + ((u >> 16) & 1u);   // round-to-nearest-even
  return (unsigned short)(r >> 16);
}

// ---- fp32 -> bf16 (row-major, used for A matrices) -------------------------
__global__ void k_convert_bf16(const float* __restrict__ in,
                               unsigned short* __restrict__ out, int n) {
  int i = blockIdx.x * blockDim.x + threadIdx.x;
  if (i < n) out[i] = f32_to_bf16(in[i]);
}

// ---- W (256x256 f32, row-major K x N) -> bf16 in WMMA B-fragment order -----
// Flat index o = (((ktile*16 + ntile)*32 + lane)*16 + e); element is
// W[K][N] with K = ktile*32 + (lane>=16 ? 8 : 0) + e + (e>=8 ? 8 : 0), N = ntile*16 + (lane&15).
__global__ void k_swizzle_w(const float* __restrict__ W,
                            unsigned short* __restrict__ wswz) {
  int o    = blockIdx.x * blockDim.x + threadIdx.x;   // 0..65535
  int t    = o >> 13;
  int rem  = o & 8191;
  int n    = rem >> 9;
  int rem2 = rem & 511;
  int l    = rem2 >> 4;
  int e    = rem2 & 15;
  int K = t * 32 + ((l >= 16) ? 8 : 0) + e + ((e >= 8) ? 8 : 0);
  int N = n * 16 + (l & 15);
  wswz[o] = f32_to_bf16(W[K * HID + N]);
}

// ---- degree + inv sqrt -----------------------------------------------------
__global__ void k_degree(const int* __restrict__ dst, float* __restrict__ deg,
                         int E) {
  int e = blockIdx.x * blockDim.x + threadIdx.x;
  if (e < E) atomicAdd(&deg[dst[e]], 1.0f);
}

__global__ void k_inv_sqrt(float* __restrict__ deg, int N) {
  int i = blockIdx.x * blockDim.x + threadIdx.x;
  if (i < N) deg[i] = rsqrtf(deg[i] + 1.0f);
}

// ---- WMMA GEMM: C(MxHID, f32) = A(MxHID, bf16 row-major) @ Bswz ------------
// One wave per 16x16 C tile; 8 K-iterations of v_wmma_f32_16x16x32_bf16.
// W fragments are pre-swizzled so each lane reads one contiguous 32B chunk.
__global__ __launch_bounds__(256)
void k_gemm_bf16(const unsigned short* __restrict__ A,
                 const unsigned short* __restrict__ Bswz,
                 float* __restrict__ C, int M) {
  int lane  = threadIdx.x & 31;
  int wid   = blockIdx.x * 8 + (threadIdx.x >> 5);
  int mtile = wid >> 4;          // HID/16 == 16 N-tiles
  int ntile = wid & 15;
  if (mtile * 16 >= M) return;   // wave-uniform: EXEC stays all-ones

  const uvec4* A4 = (const uvec4*)A;
  const uvec4* B4 = (const uvec4*)Bswz;

  int m = mtile * 16 + (lane & 15);
  if (m >= M) m = M - 1;                       // keep all lanes active
  int aoff = (lane & 16) ? 8 : 0;              // K sub-offset per lane group

  v8f acc = {};
#pragma unroll
  for (int t = 0; t < 8; ++t) {
    ABFrag a, b;
    int abase = (m * HID + t * 32 + aoff) >> 3;        // uvec4 units
    a.u[0] = A4[abase];                                // K [koff, koff+8)
    a.u[1] = A4[abase + 2];                            // K [koff+16, koff+24)
    int bbase = (((t * 16 + ntile) * 32) + lane) << 1; // uvec4 units
    b.u[0] = B4[bbase];
    b.u[1] = B4[bbase + 1];
    acc = __builtin_amdgcn_wmma_f32_16x16x32_bf16(
        false, a.v, false, b.v, (short)0, acc, false, false);
  }

  int row0 = mtile * 16 + ((lane & 16) ? 8 : 0);
  int col  = ntile * 16 + (lane & 15);
#pragma unroll
  for (int r = 0; r < 8; ++r) {
    int row = row0 + r;
    if (row < M) C[row * HID + col] = acc[r];
  }
}

// ---- edge scatter: agg[dst] += hw[src] * inv[src]*inv[dst] -----------------
// 64 threads per edge, float4 gather (L2-resident hw) + 4 f32 atomics.
__global__ void k_scatter(const float* __restrict__ hw,
                          const int* __restrict__ src,
                          const int* __restrict__ dst,
                          const float* __restrict__ inv,
                          float* __restrict__ agg, int E) {
  int tid = blockIdx.x * blockDim.x + threadIdx.x;
  int e = tid >> 6;
  if (e >= E) return;
  int q = tid & 63;
  int s = src[e], d = dst[e];
  float coef = inv[s] * inv[d];
  fvec4 v = ((const fvec4*)(hw + (size_t)s * HID))[q];
  float* ap = agg + (size_t)d * HID + q * 4;
  atomicAdd(ap + 0, v.x * coef);
  atomicAdd(ap + 1, v.y * coef);
  atomicAdd(ap + 2, v.z * coef);
  atomicAdd(ap + 3, v.w * coef);
}

// ---- combine: relu(agg + hw*inv^2 + bias); optional bf16 / f32 outputs -----
__global__ void k_combine(const float* __restrict__ hw,
                          const float* __restrict__ agg,
                          const float* __restrict__ inv,
                          const float* __restrict__ bias,
                          unsigned short* __restrict__ out_bf,
                          float* __restrict__ out_f32, int total) {
  int i = blockIdx.x * blockDim.x + threadIdx.x;
  if (i >= total) return;
  int node = i >> 8;
  int f    = i & 255;
  float iv = inv[node];
  float v  = fmaf(hw[i], iv * iv, agg[i]) + bias[f];
  v = fmaxf(v, 0.0f);
  if (out_bf)  out_bf[i]  = f32_to_bf16(v);
  if (out_f32) out_f32[i] = v;
}

// ---- pooling ---------------------------------------------------------------
__global__ void k_count(const int* __restrict__ batch,
                        float* __restrict__ counts, int N) {
  int i = blockIdx.x * blockDim.x + threadIdx.x;
  if (i < N) atomicAdd(&counts[batch[i]], 1.0f);
}

__global__ void k_pool(const float* __restrict__ emb,
                       const int* __restrict__ batch,
                       float* __restrict__ sums, int total) {
  int i = blockIdx.x * blockDim.x + threadIdx.x;
  if (i >= total) return;
  int node = i >> 8;
  int f    = i & 255;
  atomicAdd(&sums[batch[node] * HID + f], emb[i]);
}

__global__ void k_mean(const float* __restrict__ sums,
                       const float* __restrict__ counts,
                       float* __restrict__ gemb) {
  int i = blockIdx.x * blockDim.x + threadIdx.x;   // GNUM*HID threads
  gemb[i] = sums[i] / fmaxf(counts[i >> 8], 1.0f);
}

// ---- classifier MLP (tiny: VALU is fine) -----------------------------------
__global__ void k_fc1(const float* __restrict__ gemb,
                      const float* __restrict__ Wc1,
                      const float* __restrict__ bc1,
                      float* __restrict__ hcls) {
  __shared__ float row[HID];
  int g = blockIdx.x;       // GNUM blocks
  int t = threadIdx.x;      // 128 threads
  row[t]       = gemb[g * HID + t];
  row[t + 128] = gemb[g * HID + t + 128];
  __syncthreads();
  float s = bc1[t];
#pragma unroll 4
  for (int k = 0; k < HID; ++k) s = fmaf(row[k], Wc1[k * 128 + t], s);
  hcls[g * 128 + t] = fmaxf(s, 0.0f);
}

__global__ void k_fc2(const float* __restrict__ hcls,
                      const float* __restrict__ Wc2,
                      const float* __restrict__ bc2,
                      float* __restrict__ logits) {
  int i = blockIdx.x * blockDim.x + threadIdx.x;
  if (i >= GNUM * 2) return;
  int g = i >> 1, c = i & 1;
  float s = bc2[c];
#pragma unroll 4
  for (int k = 0; k < 128; ++k) s = fmaf(hcls[g * 128 + k], Wc2[k * 2 + c], s);
  logits[i] = s;
}

// ---------------------------------------------------------------------------
extern "C" void kernel_launch(void* const* d_in, const int* in_sizes, int n_in,
                              void* d_out, int out_size, void* d_ws,
                              size_t ws_size, hipStream_t stream) {
  (void)n_in; (void)out_size; (void)ws_size;
  const float* x    = (const float*)d_in[0];
  const int*   eidx = (const int*)d_in[1];
  const int*   batch= (const int*)d_in[2];
  const float* W1   = (const float*)d_in[4];
  const float* b1   = (const float*)d_in[5];
  const float* W2   = (const float*)d_in[6];
  const float* b2   = (const float*)d_in[7];
  const float* Wc1  = (const float*)d_in[8];
  const float* bc1  = (const float*)d_in[9];
  const float* Wc2  = (const float*)d_in[10];
  const float* bc2  = (const float*)d_in[11];

  const int N = in_sizes[0] / HID;
  const int E = in_sizes[1] / 2;
  const int* src  = eidx;
  const int* dstp = eidx + E;

  float* out      = (float*)d_out;
  float* logits   = out;                         // (G,2) first
  float* node_emb = out + (size_t)GNUM * 2;      // (N,HID) second

  // -------- workspace carve (all 256B aligned) ------------------------------
  char* ws = (char*)d_ws;
  size_t off = 0;
  auto carve = [&](size_t bytes) -> char* {
    char* p = ws + off;
    off += (bytes + 255) & ~(size_t)255;
    return p;
  };
  unsigned short* xbf = (unsigned short*)carve((size_t)N * HID * 2);
  unsigned short* hbf = (unsigned short*)carve((size_t)N * HID * 2);
  unsigned short* w1s = (unsigned short*)carve((size_t)HID * HID * 2);
  unsigned short* w2s = (unsigned short*)carve((size_t)HID * HID * 2);
  float* hw     = (float*)carve((size_t)N * HID * 4);
  float* agg    = (float*)carve((size_t)N * HID * 4);
  float* inv    = (float*)carve((size_t)N * 4);
  float* sums   = (float*)carve((size_t)GNUM * HID * 4);
  float* counts = (float*)carve((size_t)GNUM * 4);
  float* gemb   = (float*)carve((size_t)GNUM * HID * 4);
  float* hcls   = (float*)carve((size_t)GNUM * (HID / 2) * 4);

  const int total = N * HID;

  // -------- init ------------------------------------------------------------
  hipMemsetAsync(inv,    0, (size_t)N * 4,           stream);
  hipMemsetAsync(agg,    0, (size_t)total * 4,       stream);
  hipMemsetAsync(sums,   0, (size_t)GNUM * HID * 4,  stream);
  hipMemsetAsync(counts, 0, (size_t)GNUM * 4,        stream);

  k_convert_bf16<<<(total + 255) / 256, 256, 0, stream>>>(x, xbf, total);
  k_swizzle_w<<<(HID * HID) / 256, 256, 0, stream>>>(W1, w1s);
  k_swizzle_w<<<(HID * HID) / 256, 256, 0, stream>>>(W2, w2s);
  k_degree<<<(E + 255) / 256, 256, 0, stream>>>(dstp, inv, E);
  k_inv_sqrt<<<(N + 255) / 256, 256, 0, stream>>>(inv, N);

  const int mtiles     = (N + 15) / 16;
  const int gemmBlocks = (mtiles * 16 + 7) / 8;   // 8 waves/block
  const int scatBlocks = (int)(((long long)E * 64 + 255) / 256);

  // -------- layer 1 ---------------------------------------------------------
  k_gemm_bf16<<<gemmBlocks, 256, 0, stream>>>(xbf, w1s, hw, N);
  k_scatter<<<scatBlocks, 256, 0, stream>>>(hw, src, dstp, inv, agg, E);
  k_combine<<<(total + 255) / 256, 256, 0, stream>>>(
      hw, agg, inv, b1, hbf, (float*)nullptr, total);

  // -------- layer 2 ---------------------------------------------------------
  hipMemsetAsync(agg, 0, (size_t)total * 4, stream);
  k_gemm_bf16<<<gemmBlocks, 256, 0, stream>>>(hbf, w2s, hw, N);
  k_scatter<<<scatBlocks, 256, 0, stream>>>(hw, src, dstp, inv, agg, E);
  k_combine<<<(total + 255) / 256, 256, 0, stream>>>(
      hw, agg, inv, b2, (unsigned short*)nullptr, node_emb, total);

  // -------- pooling + classifier --------------------------------------------
  k_count<<<(N + 255) / 256, 256, 0, stream>>>(batch, counts, N);
  k_pool<<<(total + 255) / 256, 256, 0, stream>>>(node_emb, batch, sums, total);
  k_mean<<<(GNUM * HID) / 256, 256, 0, stream>>>(sums, counts, gemb);
  k_fc1<<<GNUM, 128, 0, stream>>>(gemb, Wc1, bc1, hcls);
  k_fc2<<<2, 256, 0, stream>>>(hcls, Wc2, bc2, logits);
}
